// ScaledDotProductAttention_20916490731906
// MI455X (gfx1250) — compile-verified, matrix operand order checked by reading, not run on hardware
//
#include <hip/hip_runtime.h>

typedef __attribute__((ext_vector_type(16))) _Float16 v16h;
typedef __attribute__((ext_vector_type(8)))  float    v8f;

#define B_ 8
#define S_ 1024
#define H_ 16
#define D_ 64
#define M_ 1024

// K-index inside a 16x32 f16 A/B fragment for element j of the v16h,
// given this lane-half's base (0 for lanes 0-15, 8 for lanes 16-31).
// VGPR i (pair j=2i,2i+1): K = base + 2i (i<4)  or  base + 16 + 2(i-4).
__device__ __forceinline__ int k_of(int j, int kbase) {
  int i = j >> 1;
  int base = (i < 4) ? (2 * i) : (2 * i + 8);
  return kbase + base + (j & 1);
}

// pack two f32 into one dword of two f16 (v_cvt_pk_rz_f16_f32)
__device__ __forceinline__ unsigned int pk2h(float x, float y) {
  return __builtin_bit_cast(unsigned int, __builtin_amdgcn_cvt_pkrtz(x, y));
}

// raw v_max_num_f32 (skip fmaxf's canonicalize chaff; inputs are never sNaN)
__device__ __forceinline__ float maxr(float a, float b) {
  float d;
  asm("v_max_num_f32 %0, %1, %2" : "=v"(d) : "v"(a), "v"(b));
  return d;
}

// DPP16 lane permute within each 16-lane row (VALU, no LDS traffic)
template <int CTRL>
__device__ __forceinline__ float dpp_mov(float x) {
  return __builtin_bit_cast(
      float, __builtin_amdgcn_update_dpp(0, __builtin_bit_cast(int, x), CTRL,
                                         0xF, 0xF, true));
}
#define DPP_XOR1 0xB1   // quad_perm(1,0,3,2)
#define DPP_XOR2 0x4E   // quad_perm(2,3,0,1)
#define DPP_ROR4 0x124  // row_ror:4
#define DPP_ROR8 0x128  // row_ror:8

// 16-lane all-reduce (rotation butterfly: quad stages make groups-of-4
// uniform, then ror4 + ror8 combine the four groups exactly once)
__device__ __forceinline__ float allmax16(float x) {
  x = maxr(x, dpp_mov<DPP_XOR1>(x));
  x = maxr(x, dpp_mov<DPP_XOR2>(x));
  x = maxr(x, dpp_mov<DPP_ROR4>(x));
  x = maxr(x, dpp_mov<DPP_ROR8>(x));
  return x;
}
__device__ __forceinline__ float allsum16(float x) {
  x += dpp_mov<DPP_XOR1>(x);
  x += dpp_mov<DPP_XOR2>(x);
  x += dpp_mov<DPP_ROR4>(x);
  x += dpp_mov<DPP_ROR8>(x);
  return x;
}

__global__ void zero_bits_kernel(unsigned int* bits, int n) {
  int i = blockIdx.x * blockDim.x + threadIdx.x;
  if (i < n) bits[i] = 0u;
}

__global__ void scatter_bits_kernel(const int* __restrict__ mb,
                                    const int* __restrict__ ms,
                                    unsigned int* bits, int m) {
  int i = blockIdx.x * blockDim.x + threadIdx.x;
  if (i < m) {
    int bi = mb[i] * S_ + ms[i];
    atomicOr(&bits[bi >> 5], 1u << (bi & 31));
  }
}

__launch_bounds__(128)
__global__ void flash_attn_kernel(const float* __restrict__ Qg,
                                  const float* __restrict__ Kg,
                                  const float* __restrict__ Vg,
                                  const unsigned int* __restrict__ qbits,
                                  const unsigned int* __restrict__ kbits,
                                  float* __restrict__ Og) {
  // pad rows to 68 halves (136 B): fragment reads conflict-free, 8B-aligned
  __shared__ _Float16 Kt[64][68];
  __shared__ _Float16 Vt[64][68];
  __shared__ _Float16 Pt[4][16][68];

  const int qb    = blockIdx.x & 15;   // 16 query superblocks of 64
  const int bh    = blockIdx.x >> 4;
  const int b     = bh / H_;
  const int h     = bh % H_;
  const int wave  = threadIdx.x >> 5;
  const int lane  = threadIdx.x & 31;
  const int l16   = lane & 15;
  const int hi16  = lane >> 4;         // lane half
  const int kbase = hi16 ? 8 : 0;      // K range owned by this half (A/B frags)
  const int h8    = hi16 ? 8 : 0;      // M offset owned by this half (C frags)

  const int q0 = qb * 64 + wave * 16;

  // ---- Q: two 16x32 f16 A-fragments covering D=64 (loaded once).
  // Fold scale = 1/sqrt(64) = 0.125 (power of two: exact in f16) into Q.
  v16h aq[2];
  {
    const float* qp = Qg + ((size_t)(b * S_ + q0 + l16) * H_ + h) * D_;
#pragma unroll
    for (int c = 0; c < 2; ++c)
#pragma unroll
      for (int j = 0; j < 16; ++j)
        aq[c][j] = (_Float16)(qp[c * 32 + k_of(j, kbase)] * 0.125f);
  }

  // ---- query-row mask bits for the 8 C-rows this lane accumulates ----
  bool rowm[8];
#pragma unroll
  for (int r = 0; r < 8; ++r) {
    int bi = b * S_ + q0 + r + h8;
    rowm[r] = (qbits[bi >> 5] >> (bi & 31)) & 1u;
  }

  // ---- accumulators: O (16x64 as 4 C tiles), online-softmax m,l ----
  v8f o[4];
  float m_i[8], l_i[8];
#pragma unroll
  for (int c = 0; c < 4; ++c)
#pragma unroll
    for (int r = 0; r < 8; ++r) o[c][r] = 0.f;
#pragma unroll
  for (int r = 0; r < 8; ++r) { m_i[r] = -3.0e38f; l_i[r] = 0.f; }

  const float* kb_p = Kg + ((size_t)b * S_ * H_ + h) * D_;     // + s*H*D + d
  const float* vb_p = Vg + ((size_t)(b * H_ + h)) * S_ * D_;   // + s*D + d

  for (int kb = 0; kb < S_; kb += 64) {   // 64-key blocks: 16 WMMA / iter
    __syncthreads();  // prev iteration's Vt/Pt reads done before overwrite

    // cooperative stage of 64x64 K and V tiles:
    // global_load_b128 -> v_cvt_pk_rz_f16_f32 x2 -> ds_store_b64
    {
      const float4* kp4 = (const float4*)(kb_p + (size_t)kb * H_ * D_);
      const float4* vp4 = (const float4*)(vb_p + (size_t)kb * D_);
#pragma unroll
      for (int e = threadIdx.x; e < 64 * 16; e += 128) {  // 1024 float4s
        int key = e >> 4, d4 = e & 15;
        float4 kq = kp4[(size_t)key * (H_ * D_ / 4) + d4];
        float4 vq = vp4[(size_t)key * (D_ / 4) + d4];
        uint2 kw, vw;
        kw.x = pk2h(kq.x, kq.y);  kw.y = pk2h(kq.z, kq.w);
        vw.x = pk2h(vq.x, vq.y);  vw.y = pk2h(vq.z, vq.w);
        *(uint2*)&Kt[key][d4 * 4] = kw;   // 8B-aligned (row stride 136 B)
        *(uint2*)&Vt[key][d4 * 4] = vw;
      }
    }
    __syncthreads();

    // ---- scores S = (Q*scale) K^T : four 16x16 key sub-tiles ----
    v8f s[4];
#pragma unroll
    for (int t = 0; t < 4; ++t)
#pragma unroll
      for (int r = 0; r < 8; ++r) s[t][r] = 0.f;
#pragma unroll
    for (int c = 0; c < 2; ++c) {
#pragma unroll
      for (int t = 0; t < 4; ++t) {
        v16h bk;
#pragma unroll
        for (int j = 0; j < 16; ++j)
          bk[j] = Kt[t * 16 + l16][c * 32 + k_of(j, kbase)];
        s[t] = __builtin_amdgcn_wmma_f32_16x16x32_f16(false, aq[c], false, bk,
                                                      (short)0, s[t], false, false);
      }
    }

    // ---- masks (reference: set masked entries to -1e10) ----
    bool cm[4];
#pragma unroll
    for (int t = 0; t < 4; ++t) {
      int ki = b * S_ + kb + t * 16 + l16;
      cm[t] = (kbits[ki >> 5] >> (ki & 31)) & 1u;
    }
#pragma unroll
    for (int t = 0; t < 4; ++t)
#pragma unroll
      for (int r = 0; r < 8; ++r)
        if (cm[t] || rowm[r]) s[t][r] = -1.0e10f;

    // ---- online softmax (row reductions via DPP16, pure VALU) ----
#pragma unroll
    for (int r = 0; r < 8; ++r) {
      float bm = maxr(maxr(s[0][r], s[1][r]), maxr(s[2][r], s[3][r]));
      bm = allmax16(bm);
      float mnew = maxr(m_i[r], bm);
      float p0 = __expf(s[0][r] - mnew);
      float p1 = __expf(s[1][r] - mnew);
      float p2 = __expf(s[2][r] - mnew);
      float p3 = __expf(s[3][r] - mnew);
      float rs = allsum16((p0 + p1) + (p2 + p3));
      float alpha = __expf(m_i[r] - mnew);   // first block: exp(-3e38-x) = 0
      l_i[r] = l_i[r] * alpha + rs;
      m_i[r] = mnew;
#pragma unroll
      for (int c = 0; c < 4; ++c) o[c][r] *= alpha;
      // C-layout -> LDS so we can re-read P in A-fragment layout
      Pt[wave][r + h8][l16]      = (_Float16)p0;
      Pt[wave][r + h8][16 + l16] = (_Float16)p1;
      Pt[wave][r + h8][32 + l16] = (_Float16)p2;
      Pt[wave][r + h8][48 + l16] = (_Float16)p3;
    }
    __syncthreads();  // Pt visible across the wave's lane halves

    // ---- O += P @ V : K = 64 keys as two 32-key halves, 4 dim-chunks ----
    v16h ap[2];
#pragma unroll
    for (int kh = 0; kh < 2; ++kh)
#pragma unroll
      for (int j = 0; j < 16; ++j)
        ap[kh][j] = Pt[wave][l16][kh * 32 + k_of(j, kbase)];
#pragma unroll
    for (int c = 0; c < 4; ++c) {
#pragma unroll
      for (int kh = 0; kh < 2; ++kh) {
        v16h bv;
#pragma unroll
        for (int j = 0; j < 16; ++j)
          bv[j] = Vt[kh * 32 + k_of(j, kbase)][c * 16 + l16];
        o[c] = __builtin_amdgcn_wmma_f32_16x16x32_f16(false, ap[kh], false, bv,
                                                      (short)0, o[c], false, false);
      }
    }
  }

  // ---- epilogue: normalize by l and store O (B,H,S,D) fp32 ----
#pragma unroll
  for (int r = 0; r < 8; ++r) {
    float inv = 1.0f / l_i[r];
    int row = q0 + r + h8;
    float* op = Og + ((size_t)(b * H_ + h) * S_ + row) * D_;
#pragma unroll
    for (int c = 0; c < 4; ++c) op[c * 16 + l16] = o[c][r] * inv;
  }
}

extern "C" void kernel_launch(void* const* d_in, const int* in_sizes, int n_in,
                              void* d_out, int out_size, void* d_ws, size_t ws_size,
                              hipStream_t stream) {
  (void)in_sizes; (void)n_in; (void)out_size; (void)ws_size;
  const float* Q  = (const float*)d_in[0];
  const float* K  = (const float*)d_in[1];
  const float* V  = (const float*)d_in[2];
  const int* qm_b = (const int*)d_in[3];
  const int* qm_s = (const int*)d_in[4];
  const int* km_b = (const int*)d_in[5];
  const int* km_s = (const int*)d_in[6];

  unsigned int* qbits = (unsigned int*)d_ws;        // 256 words
  unsigned int* kbits = qbits + (B_ * S_ / 32);     // 256 words
  const int nwords = B_ * S_ / 32;

  zero_bits_kernel<<<(2 * nwords + 255) / 256, 256, 0, stream>>>(qbits, 2 * nwords);
  scatter_bits_kernel<<<(M_ + 255) / 256, 256, 0, stream>>>(qm_b, qm_s, qbits, M_);
  scatter_bits_kernel<<<(M_ + 255) / 256, 256, 0, stream>>>(km_b, km_s, kbits, M_);

  dim3 grid(B_ * H_ * (S_ / 64));   // 2048 workgroups
  flash_attn_kernel<<<grid, 128, 0, stream>>>(Q, K, V, qbits, kbits, (float*)d_out);
}
